// SharedCodebook3Way_56590489092792
// MI455X (gfx1250) — matrix operand order repaired and checked by reading, not compiled, not run on hardware
//
#include <hip/hip_runtime.h>
#include <hip/hip_bf16.h>

// ---------------- CDNA5 WMMA types / helpers ----------------
typedef __bf16 bf16_t;
typedef __attribute__((ext_vector_type(16))) __bf16 v16bf;
typedef __attribute__((ext_vector_type(8)))  float  v8f;

#define NVEC   65536
#define D_IN   1024
#define D_CODE 256
#define N_CODES 64

// workspace layout (see kernel_launch)
#define WS_F32_TOTAL 16520            // floats before bf16 region (16B-aligned end)

__device__ __forceinline__ v8f wmma_bf16(v16bf a, v16bf b, v8f c) {
  // D = A(16x32 bf16) * B(32x16 bf16) + C(16x16 f32)
  return __builtin_amdgcn_wmma_f32_16x16x32_bf16(
      /*neg_a=*/false, a, /*neg_b=*/false, b,
      /*c_mod=*/(short)0, c, /*reuse_a=*/false, /*reuse_b=*/false);
}

// A fragment: lane = half*16 + m ; element e -> K=16*half+e
__device__ __forceinline__ v16bf load_frag_a(const float* __restrict__ base, int half) {
  const float4* q0 = (const float4*)(base + 16 * half);       // 32B aligned
  const float4* q1 = (const float4*)(base + 16 * half + 8);
  float4 u0 = q0[0], u1 = q0[1], u2 = q1[0], u3 = q1[1];
  v16bf a;
  a[0]  = (bf16_t)u0.x; a[1]  = (bf16_t)u0.y; a[2]  = (bf16_t)u0.z; a[3]  = (bf16_t)u0.w;
  a[4]  = (bf16_t)u1.x; a[5]  = (bf16_t)u1.y; a[6]  = (bf16_t)u1.z; a[7]  = (bf16_t)u1.w;
  a[8]  = (bf16_t)u2.x; a[9]  = (bf16_t)u2.y; a[10] = (bf16_t)u2.z; a[11] = (bf16_t)u2.w;
  a[12] = (bf16_t)u3.x; a[13] = (bf16_t)u3.y; a[14] = (bf16_t)u3.z; a[15] = (bf16_t)u3.w;
  return a;
}

// B fragment from LDS laid out [n][kk] (kk contiguous): lane = khalf*16 + n ; K = khalf*16 + e
__device__ __forceinline__ v16bf load_frag_b(const bf16_t* __restrict__ p) {
  v16bf b;
#pragma unroll
  for (int e = 0; e < 16; ++e) b[e] = p[e];
  return b;
}

// ---------------- prep kernels ----------------
__global__ void k_zero_ws(float* ws, int n) {
  int i = blockIdx.x * blockDim.x + threadIdx.x;
  if (i < n) ws[i] = 0.0f;
}

// wEncT[n*1024 + k] = bf16(W_enc[k*256 + n])   (column-major for contiguous K staging)
__global__ __launch_bounds__(256) void k_prep_wenc(const float* __restrict__ W, bf16_t* __restrict__ wt) {
  int i = blockIdx.x * 256 + threadIdx.x;      // i < 256*1024
  int n = i >> 10, k = i & 1023;
  wt[i] = (bf16_t)W[(size_t)k * D_CODE + n];
}

// wDecT[n*256 + k] = bf16(W_dec[k*1024 + n])
__global__ __launch_bounds__(256) void k_prep_wdec(const float* __restrict__ W, bf16_t* __restrict__ wt) {
  int i = blockIdx.x * 256 + threadIdx.x;      // i < 1024*256
  int n = i >> 8, k = i & 255;
  wt[i] = (bf16_t)W[(size_t)k * D_IN + n];
}

// cbf = bf16(codebook) (already [n][k] row-major) ; c2[n] = ||codebook[n]||^2
__global__ __launch_bounds__(256) void k_prep_cb(const float* __restrict__ cb,
                                                 bf16_t* __restrict__ cbf, float* __restrict__ c2) {
  int i = blockIdx.x * 256 + threadIdx.x;      // i < 64*256
  cbf[i] = (bf16_t)cb[i];
  if (blockIdx.x == 0 && threadIdx.x < N_CODES) {
    float s = 0.f;
    for (int k = 0; k < D_CODE; ++k) { float v = cb[threadIdx.x * D_CODE + k]; s += v * v; }
    c2[threadIdx.x] = s;
  }
}

// ---------------- K1: encoder GEMM + bias + LayerNorm (fused) ----------------
__global__ __launch_bounds__(256) void k_enc_ln(
    const float* __restrict__ x, const bf16_t* __restrict__ wt,
    const float* __restrict__ bias, const float* __restrict__ g,
    const float* __restrict__ bvec, float* __restrict__ out_ze) {
  __shared__ __align__(16) bf16_t ldsB[D_CODE * 32];   // W chunk, [n][kk], 16 KB
  __shared__ float  ldsZ[64 * 257];                    // z tile, padded stride
  __shared__ float  ldsMean[64], ldsRstd[64];

  const int tid  = threadIdx.x;
  const int lane = tid & 31;
  const int wave = tid >> 5;
  const int sr   = wave >> 1;
  const int ch   = wave & 1;
  const int half = lane >> 4;
  const int ln16 = lane & 15;
  const int rowBase = blockIdx.x * 64 + sr * 16;

  v8f acc[8] = {};
  const float* aRow = x + (size_t)(rowBase + ln16) * D_IN;
  const bf16_t* wcol = wt + (size_t)tid * D_IN;        // column n = tid

  for (int k0 = 0; k0 < D_IN; k0 += 32) {
    __syncthreads();
    {
      const uint4* src = (const uint4*)(wcol + k0);    // 64B contiguous
      uint4* dst = (uint4*)(&ldsB[tid * 32]);
      dst[0] = src[0]; dst[1] = src[1]; dst[2] = src[2]; dst[3] = src[3];
    }
    __syncthreads();

    v16bf afrag = load_frag_a(aRow + k0, half);
#pragma unroll
    for (int t = 0; t < 8; ++t) {
      int n = ch * 128 + t * 16 + ln16;
      v16bf bfrag = load_frag_b(&ldsB[n * 32 + half * 16]);
      acc[t] = wmma_bf16(afrag, bfrag, acc[t]);
    }
  }

#pragma unroll
  for (int t = 0; t < 8; ++t) {
    int col = ch * 128 + t * 16 + ln16;
    float bb = bias[col];
#pragma unroll
    for (int r = 0; r < 8; ++r) {
      int rowl = sr * 16 + r + 8 * half;
      ldsZ[rowl * 257 + col] = acc[t][r] + bb;
    }
  }
  __syncthreads();

  if (tid < 64) {
    float s = 0.f, ss = 0.f;
    for (int c = 0; c < D_CODE; ++c) { float v = ldsZ[tid * 257 + c]; s += v; ss += v * v; }
    float mean = s * (1.0f / D_CODE);
    float var  = ss * (1.0f / D_CODE) - mean * mean;
    ldsMean[tid] = mean;
    ldsRstd[tid] = rsqrtf(var + 1e-5f);
  }
  __syncthreads();

  float gc = g[tid], bc = bvec[tid];
  for (int i = 0; i < 64; ++i) {
    float v  = ldsZ[i * 257 + tid];
    float ze = (v - ldsMean[i]) * ldsRstd[i] * gc + bc;
    out_ze[(size_t)(blockIdx.x * 64 + i) * D_CODE + tid] = ze;
  }
}

// ---------------- K2: VQ argmin + gather + EMA scatter ----------------
__global__ __launch_bounds__(256) void k_vq(
    const float* __restrict__ ze, const float* __restrict__ codebook,
    const bf16_t* __restrict__ cbf, const float* __restrict__ c2,
    float* __restrict__ out_idx, float* __restrict__ out_zq,
    float* ws_counts, float* ws_embed, float* ws_commit) {
  __shared__ __align__(16) bf16_t ldsB[N_CODES * 32];  // codebook chunk [n][kk]
  __shared__ float  ldsS[64 * 68];                     // scores, padded
  __shared__ float  ldsC2[N_CODES];
  __shared__ int    ldsIdx[64];

  const int tid  = threadIdx.x;
  const int lane = tid & 31;
  const int wave = tid >> 5;
  const int sr   = wave >> 1;
  const int ch   = wave & 1;
  const int half = lane >> 4;
  const int ln16 = lane & 15;
  const int rowBase = blockIdx.x * 64 + sr * 16;

  if (tid < N_CODES) ldsC2[tid] = c2[tid];

  v8f acc[2] = {};
  const float* aRow = ze + (size_t)(rowBase + ln16) * D_CODE;

  for (int k0 = 0; k0 < D_CODE; k0 += 32) {
    __syncthreads();
    {
      int n = tid >> 2, ks = (tid & 3) * 8;            // 16B per thread
      *(uint4*)(&ldsB[n * 32 + ks]) = *(const uint4*)(cbf + n * D_CODE + k0 + ks);
    }
    __syncthreads();
    v16bf afrag = load_frag_a(aRow + k0, half);
#pragma unroll
    for (int t = 0; t < 2; ++t) {
      int n = ch * 32 + t * 16 + ln16;
      v16bf bfrag = load_frag_b(&ldsB[n * 32 + half * 16]);
      acc[t] = wmma_bf16(afrag, bfrag, acc[t]);
    }
  }
  __syncthreads();
#pragma unroll
  for (int t = 0; t < 2; ++t) {
    int col = ch * 32 + t * 16 + ln16;
#pragma unroll
    for (int r = 0; r < 8; ++r)
      ldsS[(sr * 16 + r + 8 * half) * 68 + col] = acc[t][r];
  }
  __syncthreads();

  if (tid < 64) {                                      // ||z||^2 term is row-constant
    float best = 3.402823466e+38f; int bi = 0;
    for (int c = 0; c < N_CODES; ++c) {
      float d = ldsC2[c] - 2.0f * ldsS[tid * 68 + c];
      if (d < best) { best = d; bi = c; }
    }
    ldsIdx[tid] = bi;
    out_idx[blockIdx.x * 64 + tid] = (float)bi;
    atomicAdd(&ws_counts[bi], 1.0f);
  }
  __syncthreads();

  float commit = 0.f;
  for (int i = 0; i < 64; ++i) {
    int bi = ldsIdx[i];
    size_t row = (size_t)(blockIdx.x * 64 + i);
    float zev = ze[row * D_CODE + tid];
    float zq  = codebook[bi * D_CODE + tid];
    out_zq[row * D_CODE + tid] = zq;
    float d = zev - zq; commit += d * d;
    atomicAdd(&ws_embed[bi * D_CODE + tid], zev);
  }
#pragma unroll
  for (int off = 16; off > 0; off >>= 1) commit += __shfl_xor(commit, off, 32);
  if (lane == 0) atomicAdd(ws_commit, commit);
}

// ---------------- K3: EMA finalize ----------------
__global__ __launch_bounds__(256) void k_finalize(
    const float* __restrict__ cluster_size, const float* __restrict__ ema_sum,
    const float* ws_counts, const float* ws_embed, const float* ws_commit,
    float* out_loss, float* out_ncb, float* out_ncs, float* out_nes) {
  __shared__ float ncs_s[N_CODES];
  __shared__ float nTot;
  int tid = threadIdx.x;
  if (tid < N_CODES) {
    float ncs = cluster_size[tid] * 0.99f + 0.01f * ws_counts[tid];
    out_ncs[tid] = ncs; ncs_s[tid] = ncs;
  }
  __syncthreads();
  if (tid == 0) {
    float s = 0.f;
    for (int c = 0; c < N_CODES; ++c) s += ncs_s[c];
    nTot = s;
    out_loss[0] = ws_commit[0] / ((float)NVEC * (float)D_CODE);
  }
  __syncthreads();
  float n = nTot;
  for (int i = tid; i < N_CODES * D_CODE; i += 256) {
    int c = i >> 8;
    float nes = ema_sum[i] * 0.99f + 0.01f * ws_embed[i];
    out_nes[i] = nes;
    float cs = (ncs_s[c] + 1e-5f) / (n + N_CODES * 1e-5f) * n;
    out_ncb[i] = nes / cs;
  }
}

// ---------------- K4: decoder GEMM x_recon = z_q @ W_dec + b_dec ----------------
__global__ __launch_bounds__(256) void k_dec(
    const float* __restrict__ zq, const bf16_t* __restrict__ wt,
    const float* __restrict__ bias, float* __restrict__ out_x) {
  __shared__ __align__(16) bf16_t ldsB[256 * 32];      // W_dec chunk [n][kk]

  const int tid  = threadIdx.x;
  const int lane = tid & 31;
  const int wave = tid >> 5;
  const int sr   = wave >> 1;
  const int ch   = wave & 1;
  const int half = lane >> 4;
  const int ln16 = lane & 15;
  const int rb = blockIdx.x >> 2;
  const int cb = blockIdx.x & 3;
  const int rowBase = rb * 64 + sr * 16;
  const int colBase = cb * 256;

  v8f acc[8] = {};
  const float* aRow = zq + (size_t)(rowBase + ln16) * D_CODE;
  const bf16_t* wcol = wt + (size_t)(colBase + tid) * D_CODE; // column n

  for (int k0 = 0; k0 < D_CODE; k0 += 32) {
    __syncthreads();
    {
      const uint4* src = (const uint4*)(wcol + k0);
      uint4* dst = (uint4*)(&ldsB[tid * 32]);
      dst[0] = src[0]; dst[1] = src[1]; dst[2] = src[2]; dst[3] = src[3];
    }
    __syncthreads();

    v16bf afrag = load_frag_a(aRow + k0, half);
#pragma unroll
    for (int t = 0; t < 8; ++t) {
      int n = ch * 128 + t * 16 + ln16;
      v16bf bfrag = load_frag_b(&ldsB[n * 32 + half * 16]);
      acc[t] = wmma_bf16(afrag, bfrag, acc[t]);
    }
  }

#pragma unroll
  for (int t = 0; t < 8; ++t) {
    int n = colBase + ch * 128 + t * 16 + ln16;
    float bb = bias[n];
#pragma unroll
    for (int r = 0; r < 8; ++r) {
      size_t row = (size_t)(rowBase + r + 8 * half);
      out_x[row * D_IN + n] = acc[t][r] + bb;
    }
  }
}

// ---------------- host-side launch ----------------
extern "C" void kernel_launch(void* const* d_in, const int* in_sizes, int n_in,
                              void* d_out, int out_size, void* d_ws, size_t ws_size,
                              hipStream_t stream) {
  const float* x            = (const float*)d_in[0];
  const float* W_enc        = (const float*)d_in[1];
  const float* b_enc        = (const float*)d_in[2];
  const float* ln_g         = (const float*)d_in[3];
  const float* ln_b         = (const float*)d_in[4];
  const float* codebook     = (const float*)d_in[5];
  const float* cluster_size = (const float*)d_in[6];
  const float* ema_sum      = (const float*)d_in[7];
  const float* W_dec        = (const float*)d_in[8];
  const float* b_dec        = (const float*)d_in[9];

  float* out = (float*)d_out;
  float* o_xrecon = out;                                   // [N, D_IN]
  float* o_loss   = o_xrecon + (size_t)NVEC * D_IN;        // [1]
  float* o_idx    = o_loss + 1;                            // [N]
  float* o_ze     = o_idx + NVEC;                          // [N, D_CODE]
  float* o_zq     = o_ze + (size_t)NVEC * D_CODE;          // [N, D_CODE]
  float* o_ncb    = o_zq + (size_t)NVEC * D_CODE;          // [N_CODES, D_CODE]
  float* o_ncs    = o_ncb + N_CODES * D_CODE;              // [N_CODES]
  float* o_nes    = o_ncs + N_CODES;                       // [N_CODES, D_CODE]

  // workspace: f32 atomics/constants, then bf16 weight/codebook copies
  float* ws        = (float*)d_ws;
  float* ws_counts = ws;                                   // [64]
  float* ws_embed  = ws + 64;                              // [64*256]
  float* ws_commit = ws + 16448;                           // [1]
  float* ws_c2     = ws + 16452;                           // [64]
  bf16_t* wEncT = (bf16_t*)((char*)d_ws + (size_t)WS_F32_TOTAL * 4); // [256][1024]
  bf16_t* wDecT = wEncT + (size_t)D_CODE * D_IN;           // [1024][256]
  bf16_t* cbf   = wDecT + (size_t)D_IN * D_CODE;           // [64][256]

  k_zero_ws<<<(16449 + 255) / 256, 256, 0, stream>>>(ws, 16449);
  k_prep_wenc<<<(D_CODE * D_IN) / 256, 256, 0, stream>>>(W_enc, wEncT);
  k_prep_wdec<<<(D_IN * D_CODE) / 256, 256, 0, stream>>>(W_dec, wDecT);
  k_prep_cb<<<(N_CODES * D_CODE) / 256, 256, 0, stream>>>(codebook, cbf, ws_c2);

  k_enc_ln<<<NVEC / 64, 256, 0, stream>>>(x, wEncT, b_enc, ln_g, ln_b, o_ze);
  k_vq<<<NVEC / 64, 256, 0, stream>>>(o_ze, codebook, cbf, ws_c2, o_idx, o_zq,
                                      ws_counts, ws_embed, ws_commit);
  k_finalize<<<1, 256, 0, stream>>>(cluster_size, ema_sum, ws_counts, ws_embed,
                                    ws_commit, o_loss, o_ncb, o_ncs, o_nes);
  k_dec<<<(NVEC / 64) * 4, 256, 0, stream>>>(o_zq, wDecT, b_dec, o_xrecon);
}
